// MHA_2456721293697
// MI455X (gfx1250) — matrix operand order repaired and checked by reading, not compiled, not run on hardware
//
#include <hip/hip_runtime.h>

// ---------------------------------------------------------------------------
// MHA forward for MI455X (gfx1250, wave32, WMMA + TDM).
//   B=2, S=2048, D=1024, H=16, DK=64
// Pipeline:
//   1) Q/K/V projections: bf16 WMMA GEMM (f32 in -> bf16 out to workspace)
//   2) flash-attention (streaming softmax, transposed WMMA formulation,
//      K-tile staged via Tensor Data Mover with hardware LDS padding)
//   3) output projection: bf16 WMMA GEMM (bf16 in -> f32 out to d_out)
// ---------------------------------------------------------------------------

#define kB  2
#define kS  2048
#define kD  1024
#define kH  16
#define kDK 64

typedef __attribute__((ext_vector_type(16))) __bf16 v16bf;
typedef __attribute__((ext_vector_type(8)))  __bf16 v8bf;
typedef __attribute__((ext_vector_type(8)))  float  v8f;
typedef __attribute__((ext_vector_type(4)))  unsigned int u32x4;
typedef __attribute__((ext_vector_type(4)))  int i32x4;
typedef __attribute__((ext_vector_type(8)))  int i32x8;

#if defined(__has_builtin)
#if __has_builtin(__builtin_amdgcn_tensor_load_to_lds) && \
    __has_builtin(__builtin_amdgcn_s_wait_tensorcnt)
#define HAVE_TDM 1
#endif
#endif
#ifndef HAVE_TDM
#define HAVE_TDM 0
#endif

// ---- scalar conversions (explicit bit ops; no reliance on __bf16 arithmetic)
__device__ __forceinline__ __bf16 to_bf16(float f) {
    union { unsigned short u; __bf16 b; } cv;
    unsigned x = __float_as_uint(f);
    cv.u = (unsigned short)((x + 0x7FFFu + ((x >> 16) & 1u)) >> 16);  // RNE
    return cv.b;
}
__device__ __forceinline__ float bf16_to_f(__bf16 h) {
    union { __bf16 b; unsigned short u; } cv; cv.b = h;
    return __uint_as_float(((unsigned)cv.u) << 16);
}
__device__ __forceinline__ float ld_f(const float*  p) { return *p; }
__device__ __forceinline__ float ld_f(const __bf16* p) { return bf16_to_f(*p); }

// ---- WMMA wrapper: D = A(16x32 bf16) * B(32x16 bf16) + C(16x16 f32)
__device__ __forceinline__ v8f wmma_bf16(v16bf a, v16bf b, v8f c) {
    return __builtin_amdgcn_wmma_f32_16x16x32_bf16(
        /*neg_a=*/false, a, /*neg_b=*/false, b,
        /*c_mod=*/(short)0, c, /*reuse_a=*/false, /*reuse_b=*/false);
}

// ---- assemble a 16-element bf16 fragment from two 8-half (16B) runs
__device__ __forceinline__ v16bf frag16(const __bf16* p0, const __bf16* p1) {
    v8bf lo = *(const v8bf*)p0;
    v8bf hi = *(const v8bf*)p1;
    v16bf r;
#pragma unroll
    for (int i = 0; i < 8; ++i) { r[i] = lo[i]; r[i + 8] = hi[i]; }
    return r;
}
__device__ __forceinline__ v8f zero8() {
    v8f z;
#pragma unroll
    for (int i = 0; i < 8; ++i) z[i] = 0.0f;
    return z;
}

#if HAVE_TDM
// ---- Tensor Data Mover: 2D tile global->LDS, descriptor per ISA 8.3/8.4.
// Copies tile_d1 rows x tile_d0 elements (2B each) from a row-major tensor
// (row stride = tens_stride elems) into LDS, inserting pad_amount(=4 DW)
// after every 32 DW so the LDS image lands in a bank-conflict-free layout.
__device__ __forceinline__ void tdm_load_tile_bf16(
    unsigned lds_addr, const __bf16* gaddr,
    int tens_d0, int tens_d1, int tens_stride,
    int tile_d0, int tile_d1)
{
    const unsigned long long ga = (unsigned long long)(size_t)gaddr;
    u32x4 g0;
    g0[0] = 1u;                                         // count=1 (valid user D#)
    g0[1] = lds_addr;                                   // lds_addr [63:32]
    g0[2] = (unsigned)(ga & 0xFFFFFFFFu);               // global_addr[31:0]
    g0[3] = (unsigned)((ga >> 32) & 0x01FFFFFFu)        // global_addr[56:32]
          | (2u << 30);                                 // type=2 ("image")
    i32x8 g1;
    g1[0] = (1 << 16)        // data_size = 1 -> 2 bytes
          | (1 << 20)        // pad_enable
          | (4 << 22)        // pad_interval: 32 DWORDs (128 B)
          | (3 << 25);       // pad_amount:   4 DWORDs (16 B)
    g1[1] = (tens_d0 & 0xFFFF) << 16;                   // tensor_dim0[15:0]
    g1[2] = ((unsigned)tens_d0 >> 16)                   // tensor_dim0[31:16]
          | ((tens_d1 & 0xFFFF) << 16);                 // tensor_dim1[15:0]
    g1[3] = ((unsigned)tens_d1 >> 16)                   // tensor_dim1[31:16]
          | ((tile_d0 & 0xFFFF) << 16);                 // tile_dim0
    g1[4] = tile_d1 & 0xFFFF;                           // tile_dim1 (tile_dim2=0)
    g1[5] = tens_stride;                                // tensor_dim0_stride[31:0]
    g1[6] = 0;                                          // stride hi / dim1_stride lo
    g1[7] = 0;
    i32x4 z4 = {0, 0, 0, 0};
#if __clang_major__ >= 23
    i32x8 z8 = {0, 0, 0, 0, 0, 0, 0, 0};
    __builtin_amdgcn_tensor_load_to_lds(g0, g1, z4, z4, z8, 0);
#else
    __builtin_amdgcn_tensor_load_to_lds(g0, g1, z4, z4, 0);
#endif
}
#endif

// ===========================================================================
// GEMM: C[M,N] = A[M,K] * W[N,K]^T + bias[N]     (== nn.Linear)
// block 256 thr = 8 waves; block tile 128x128, BK=32; wave tile 32x64.
// A/W staged into LDS as bf16 with +8 half padding (stride 40 halfs = 20
// DWORDs: 16 consecutive rows map to 16 distinct banks; 16B aligned b128).
// ===========================================================================
template <typename TA, bool OUT_BF16>
__global__ __launch_bounds__(256) void gemm_bias(
    const TA* __restrict__ A, const float* __restrict__ W,
    const float* __restrict__ bias, __bf16* __restrict__ Cb,
    float* __restrict__ Cf, int M, int N, int Kd)
{
    __shared__ __bf16 As[128][40];
    __shared__ __bf16 Ws[128][40];

    const int t    = threadIdx.x;
    const int lane = t & 31;
    const int wv   = t >> 5;
    const int wm   = (wv & 3) * 32;   // 4 waves along M
    const int wn   = (wv >> 2) * 64;  // 2 waves along N
    const int lr   = lane & 15;       // fragment row/col index
    const int hh   = lane >> 4;       // lane half

    const int m0 = blockIdx.y * 128;
    const int n0 = blockIdx.x * 128;

    v8f acc[2][4];
#pragma unroll
    for (int mi = 0; mi < 2; ++mi)
#pragma unroll
        for (int ni = 0; ni < 4; ++ni) acc[mi][ni] = zero8();

    for (int k0 = 0; k0 < Kd; k0 += 32) {
        // ---- stage A (128x32) and W (128x32) as bf16
#pragma unroll
        for (int i = 0; i < 2; ++i) {
            const int slot = t + i * 256;      // 512 slots x 8 halfs
            const int row  = slot >> 2;
            const int kc   = (slot & 3) * 8;
            const TA*    sa = A + (size_t)(m0 + row) * Kd + k0 + kc;
            const float* sw = W + (size_t)(n0 + row) * Kd + k0 + kc;
#pragma unroll
            for (int j = 0; j < 8; ++j) {
                As[row][kc + j] = to_bf16(ld_f(sa + j));
                Ws[row][kc + j] = to_bf16(sw[j]);
            }
        }
        if (k0 + 32 < Kd) {   // hint next tiles into cache (global_prefetch_b8)
            __builtin_prefetch(A + (size_t)(m0 + (t >> 1)) * Kd + k0 + 32, 0, 1);
            __builtin_prefetch(W + (size_t)(n0 + (t >> 1)) * Kd + k0 + 32, 0, 1);
        }
        __syncthreads();

        // ---- fragments + 8 WMMAs per wave per k-step
        v16bf af[2], bf[4];
#pragma unroll
        for (int mi = 0; mi < 2; ++mi) {
            const __bf16* r = &As[wm + mi * 16 + lr][0];
            af[mi] = frag16(r + hh * 8, r + 16 + hh * 8);
        }
#pragma unroll
        for (int ni = 0; ni < 4; ++ni) {
            const __bf16* r = &Ws[wn + ni * 16 + lr][0];
            bf[ni] = frag16(r + hh * 8, r + 16 + hh * 8);
        }
#pragma unroll
        for (int mi = 0; mi < 2; ++mi)
#pragma unroll
            for (int ni = 0; ni < 4; ++ni)
                acc[mi][ni] = wmma_bf16(af[mi], bf[ni], acc[mi][ni]);
        __syncthreads();
    }

    // ---- epilogue: +bias, store (C layout: lane col n=lr, rows hh*8+r)
#pragma unroll
    for (int mi = 0; mi < 2; ++mi) {
#pragma unroll
        for (int ni = 0; ni < 4; ++ni) {
            const int n  = n0 + wn + ni * 16 + lr;
            const float bv = bias[n];
#pragma unroll
            for (int r = 0; r < 8; ++r) {
                const int m = m0 + wm + mi * 16 + hh * 8 + r;
                const float val = acc[mi][ni][r] + bv;
                if (OUT_BF16) Cb[(size_t)m * N + n] = to_bf16(val);
                else          Cf[(size_t)m * N + n] = val;
            }
        }
    }
}

// ===========================================================================
// Flash attention, transposed WMMA formulation.
//  S^T(key x q) = K * Q^T ;  O^T(dk x q) = V^T * P^T
//  => D-matrix column == query per lane => softmax rescale is per-lane scalar.
// block = 256 thr = 8 waves, one (b,h), 8 query tiles of 16 rows (128 q/blk).
// Streams keys in tiles of 32. K tile is DMA'd by the Tensor Data Mover
// (wave 0 issues; TDM pad feature writes the bank-conflict-free layout) while
// all 8 waves stage the transposed V tile; sync = s_wait_tensorcnt + barrier.
// ===========================================================================
__global__ __launch_bounds__(256) void attn_kernel(
    const __bf16* __restrict__ Qp, const __bf16* __restrict__ Kp,
    const __bf16* __restrict__ Vp, const int* __restrict__ mask,
    __bf16* __restrict__ ctx)
{
    __shared__ __bf16 Kt[32][72];   // K tile [key][dk], stride 36 DW: no conflicts
    __shared__ __bf16 Vt[64][40];   // V^T tile [dk][key], stride 20 DW

    const int b    = blockIdx.z;
    const int h    = blockIdx.y;
    const int t    = threadIdx.x;
    const int lane = t & 31;
    const int wv   = t >> 5;
    const int lr   = lane & 15;
    const int hh   = lane >> 4;
    const int q    = blockIdx.x * 128 + wv * 16 + lr;   // this lane's query col

    // Q B-fragments (fixed for whole loop): B[dk][q] = Q[q][dk]
    const __bf16* qrow = Qp + ((size_t)(b * kS) + q) * kD + h * kDK;
    v16bf bq[2];
    bq[0] = frag16(qrow + hh * 8,      qrow + 16 + hh * 8);        // dk 0..31
    bq[1] = frag16(qrow + 32 + hh * 8, qrow + 48 + hh * 8);        // dk 32..63

#if HAVE_TDM
    const unsigned ldsK = (unsigned)(size_t)&Kt[0][0];  // generic low 32 = LDS addr
#endif

    v8f o[4];
#pragma unroll
    for (int od = 0; od < 4; ++od) o[od] = zero8();
    float m_run = -1e30f, l_run = 0.0f;

    for (int kt = 0; kt < kS; kt += 32) {
        // ---- K tile: TDM async DMA (wave 0 only; EXEC-independent per wave)
#if HAVE_TDM
        if (wv == 0) {
            tdm_load_tile_bf16(ldsK,
                               Kp + ((size_t)(b * kS) + kt) * kD + h * kDK,
                               kD, kB * kS, kD, /*tile*/ kDK, 32);
        }
#else
        {
            const int row = t >> 3;            // 32 rows, 8 thr/row
            const int c   = (t & 7) * 8;
            const __bf16* src = Kp + ((size_t)(b * kS) + kt + row) * kD + h * kDK + c;
            *(v8bf*)&Kt[row][c] = *(const v8bf*)src;
        }
#endif
        // ---- V tile: cooperative transpose staging (overlaps the TDM)
        {
            const int key = t & 31;
            const int dk0 = (t >> 5) * 8;
            const __bf16* src = Vp + ((size_t)(b * kS) + kt + key) * kD + h * kDK + dk0;
#pragma unroll
            for (int j = 0; j < 8; ++j) Vt[dk0 + j][key] = src[j];
        }
#if HAVE_TDM
        if (wv == 0) __builtin_amdgcn_s_wait_tensorcnt(0);   // TENSORcnt==0
#endif
        __syncthreads();

        // ---- S^T tile (32 keys x 16 q): 4 WMMAs
        v8f st[2];
#pragma unroll
        for (int kh = 0; kh < 2; ++kh) {
            const __bf16* kr = &Kt[kh * 16 + lr][0];
            v16bf a0 = frag16(kr + hh * 8,      kr + 16 + hh * 8);  // dk 0..31
            v16bf a1 = frag16(kr + 32 + hh * 8, kr + 48 + hh * 8);  // dk 32..63
            v8f z = zero8();
            z = wmma_bf16(a0, bq[0], z);
            z = wmma_bf16(a1, bq[1], z);
            st[kh] = z;
        }

        // ---- scale 1/sqrt(DK) + mask (lane: fixed q, keys kt+kh*16+hh*8+r)
        float sc[2][8];
#pragma unroll
        for (int kh = 0; kh < 2; ++kh) {
            const int kb = kt + kh * 16 + hh * 8;
            const int4* mp = (const int4*)(mask + (size_t)b * kS * kS + (size_t)q * kS + kb);
            const int4 ma = mp[0], mb = mp[1];
            const int mm[8] = {ma.x, ma.y, ma.z, ma.w, mb.x, mb.y, mb.z, mb.w};
#pragma unroll
            for (int r = 0; r < 8; ++r)
                sc[kh][r] = (mm[r] == 0) ? -1e9f : st[kh][r] * 0.125f;
        }

        // ---- online softmax over the 32 keys of this tile (per query col)
        float mloc = -1e30f;
#pragma unroll
        for (int kh = 0; kh < 2; ++kh)
#pragma unroll
            for (int r = 0; r < 8; ++r) mloc = fmaxf(mloc, sc[kh][r]);
        mloc = fmaxf(mloc, __shfl_xor(mloc, 16, 32));
        const float mnew  = fmaxf(m_run, mloc);
        const float alpha = __expf(m_run - mnew);

        float p0[8], p1[8], ls = 0.0f;
#pragma unroll
        for (int r = 0; r < 8; ++r) {
            p0[r] = __expf(sc[0][r] - mnew);
            p1[r] = __expf(sc[1][r] - mnew);
            ls += p0[r] + p1[r];
        }
        ls += __shfl_xor(ls, 16, 32);
        l_run = l_run * alpha + ls;
        m_run = mnew;
#pragma unroll
        for (int od = 0; od < 4; ++od)
#pragma unroll
            for (int r = 0; r < 8; ++r) o[od][r] *= alpha;   // per-lane rescale

        // ---- re-layout P (C-layout) into B-fragment (key x q): half-swap
        v16bf pb;
#pragma unroll
        for (int r = 0; r < 8; ++r) {
            const float s0 = __shfl_xor(p0[r], 16, 32);
            const float s1 = __shfl_xor(p1[r], 16, 32);
            pb[r]     = to_bf16(hh == 0 ? p0[r] : s1);
            pb[r + 8] = to_bf16(hh == 0 ? s0 : p1[r]);
        }

        // ---- O^T += V^T * P^T : 4 WMMAs
#pragma unroll
        for (int od = 0; od < 4; ++od) {
            const __bf16* vr = &Vt[od * 16 + lr][0];
            v16bf av = frag16(vr + hh * 16, vr + hh * 16 + 8);
            o[od] = wmma_bf16(av, pb, o[od]);
        }
        __syncthreads();
    }

    // ---- normalize + store context (bf16, row-major [B,S,D]); rows of O^T
    // for fixed lane are consecutive dk => one 16B store per fragment.
    const float inv = 1.0f / l_run;
#pragma unroll
    for (int od = 0; od < 4; ++od) {
        v8bf ov;
#pragma unroll
        for (int r = 0; r < 8; ++r) ov[r] = to_bf16(o[od][r] * inv);
        __bf16* dst = ctx + ((size_t)(b * kS) + q) * kD + h * kDK + od * 16 + hh * 8;
        *(v8bf*)dst = ov;
    }
}

// ===========================================================================
extern "C" void kernel_launch(void* const* d_in, const int* in_sizes, int n_in,
                              void* d_out, int out_size, void* d_ws, size_t ws_size,
                              hipStream_t stream)
{
    const float* q    = (const float*)d_in[0];
    const float* k    = (const float*)d_in[1];
    const float* v    = (const float*)d_in[2];
    const int*   mask = (const int*)  d_in[3];
    const float* w_q  = (const float*)d_in[4];
    const float* b_q  = (const float*)d_in[5];
    const float* w_k  = (const float*)d_in[6];
    const float* b_k  = (const float*)d_in[7];
    const float* w_v  = (const float*)d_in[8];
    const float* b_v  = (const float*)d_in[9];
    const float* w_o  = (const float*)d_in[10];
    const float* b_o  = (const float*)d_in[11];

    char* ws = (char*)d_ws;
    const size_t proj = (size_t)kB * kS * kD * sizeof(__bf16);   // 8 MiB each
    __bf16* Qp = (__bf16*)(ws);
    __bf16* Kp = (__bf16*)(ws + proj);
    __bf16* Vp = (__bf16*)(ws + 2 * proj);
    __bf16* Cx = (__bf16*)(ws + 3 * proj);

    const int M = kB * kS;   // 4096
    dim3 gemm_grid(kD / 128, M / 128);   // (8, 32)

    // 1) projections (f32 in, bf16 out)
    gemm_bias<float, true><<<gemm_grid, 256, 0, stream>>>(q, w_q, b_q, Qp, nullptr, M, kD, kD);
    gemm_bias<float, true><<<gemm_grid, 256, 0, stream>>>(k, w_k, b_k, Kp, nullptr, M, kD, kD);
    gemm_bias<float, true><<<gemm_grid, 256, 0, stream>>>(v, w_v, b_v, Vp, nullptr, M, kD, kD);

    // 2) flash attention -> bf16 context
    dim3 attn_grid(kS / 128, kH, kB);    // (16, 16, 2)
    attn_kernel<<<attn_grid, 256, 0, stream>>>(Qp, Kp, Vp, mask, Cx);

    // 3) output projection (bf16 in, f32 out to d_out)
    gemm_bias<__bf16, false><<<gemm_grid, 256, 0, stream>>>(Cx, w_o, b_o, nullptr, (float*)d_out, M, kD, kD);
}
